// GCN_FeedForward_21328807592612
// MI455X (gfx1250) — matrix operand order repaired and checked by reading, not compiled, not run on hardware
//
#include <hip/hip_runtime.h>
#include <hip/hip_bf16.h>
#include <stdint.h>

// ---------------------------------------------------------------------------
// GCN feed-forward on MI455X (gfx1250).
//   h1 = relu( Din^-1/2 * scatter( (Dout^-1/2 * X) @ W1 ) + b1 )
//   h1 = dropout(h1, 0.5)
//   h2 = Din^-1/2 * scatter( (Dout^-1/2 * h1) @ W2 ) + b2
//   out = h2 @ W3 + b3
// GEMMs: v_wmma_f32_16x16x32_bf16, one wave computes a 16x(16*NT) strip with
// the A fragment held in registers across NT B fragments (4x A-traffic cut);
// all fragment loads for a K-step are batched ahead of the WMMA chain so the
// scheduler can clause them and stagger loadcnt waits.
// Aggregation stays f32 via coalesced global_atomic_add_f32 (one wave/edge).
// ---------------------------------------------------------------------------

typedef __bf16 bf16_t;
typedef __attribute__((ext_vector_type(16))) __bf16 v16bf;
typedef __attribute__((ext_vector_type(8)))  __bf16 v8bf;
typedef __attribute__((ext_vector_type(8)))  float  v8f;

#define IN_SZ  512
#define HID_SZ 256
#define OUT_SZ 16

// ---------------- elementwise helpers ----------------

__global__ void zero_f32(float* __restrict__ p, int n) {
    int i = blockIdx.x * blockDim.x + threadIdx.x;
    if (i < n) p[i] = 0.0f;
}

// degree accumulation (both endpoints in one pass)
__global__ void degree_kernel(const int* __restrict__ src, const int* __restrict__ dst,
                              float* __restrict__ deg_out, float* __restrict__ deg_in, int E) {
    int i = blockIdx.x * blockDim.x + threadIdx.x;
    if (i >= E) return;
    atomicAdd(&deg_out[src[i]], 1.0f);
    atomicAdd(&deg_in [dst[i]], 1.0f);
}

// in-place: d -> rsqrt(max(d,1))
__global__ void finalize_deg(float* __restrict__ d, int n) {
    int i = blockIdx.x * blockDim.x + threadIdx.x;
    if (i >= n) return;
    float v = d[i];
    v = v < 1.0f ? 1.0f : v;
    d[i] = rsqrtf(v);
}

// Xb[n,c] = bf16( X[n,c] * inv_out[n] )   (features pre-scaled by Dout^-1/2)
__global__ void scale_cvt_bf16(const float* __restrict__ X, const float* __restrict__ inv_out,
                               bf16_t* __restrict__ Xb, int total, int Cshift) {
    int i = blockIdx.x * blockDim.x + threadIdx.x;
    if (i >= total) return;
    int n = i >> Cshift;
    Xb[i] = (bf16_t)(X[i] * inv_out[n]);
}

// Pack row-major f32 W[K x N] into the WMMA bf16 B-operand lane layout:
//   Wp[((ntile*KC + kc)*32 + lane)*16 + i]  where for 16-bit B (32x16):
//     lanes 0-15  -> N = lane,     K = kc*32 + 0..7, 16..23
//     lanes 16-31 -> N = lane-16,  K = kc*32 + 8..15, 24..31
// so each lane's 16 values (32 bytes) are contiguous -> single vector load.
__global__ void pack_wmma_b(const float* __restrict__ W, bf16_t* __restrict__ Wp,
                            int K, int N) {
    int idx = blockIdx.x * blockDim.x + threadIdx.x;
    int total = K * N;
    if (idx >= total) return;
    int i     = idx & 15;
    int lane  = (idx >> 4) & 31;
    int rest  = idx >> 9;          // kc + ntile*KC
    int KC    = K >> 5;
    int kc    = rest % KC;
    int ntile = rest / KC;
    int n  = ntile * 16 + (lane & 15);
    int kk = kc * 32 + (i >> 3) * 16 + (lane >> 4) * 8 + (i & 7);
    Wp[idx] = (bf16_t)W[kk * N + n];
}

// ---------------- WMMA GEMM: C[M,N] = A[M,K](bf16) x Wp(packed bf16) (+bias) ----
// One wave computes a 16 x (16*NT) strip: the A fragment is loaded once per
// K-step and reused across NT accumulators. Wave-uniform guard keeps EXEC
// all-ones (WMMA requirement).
template <int NT>
__global__ void gemm_wmma_bf16(const bf16_t* __restrict__ A,
                               const bf16_t* __restrict__ Wp,
                               float* __restrict__ C,
                               const float* __restrict__ bias,
                               int M, int N, int K) {
    const int lane = threadIdx.x & 31;
    const int wave = threadIdx.x >> 5;
    const int tid  = blockIdx.x * (blockDim.x >> 5) + wave;
    const int Ngrp  = N / (16 * NT);
    const int total = (M >> 4) * Ngrp;
    if (tid >= total) return;               // uniform per wave

    const int tile_m  = tid / Ngrp;
    const int tile_ng = tid % Ngrp;         // group of NT consecutive N-tiles
    const int half    = lane >> 4;          // 0: K 0-7/16-23, 1: K 8-15/24-31
    const int mrow    = tile_m * 16 + (lane & 15);
    const int KC      = K >> 5;

    const bf16_t* __restrict__ Arow  = A + (size_t)mrow * K + half * 8;
    const v16bf*  __restrict__ Bbase =
        (const v16bf*)Wp + (size_t)(tile_ng * NT) * KC * 32 + lane;

    v8f acc[NT];
#pragma unroll
    for (int j = 0; j < NT; ++j) acc[j] = (v8f){};

    for (int kc = 0; kc < KC; ++kc) {
        // --- batch ALL fragment loads for this K-step (clause-friendly) ---
        v8bf lo = *(const v8bf*)(Arow);        // K = kA+0..7      (16B load)
        v8bf hi = *(const v8bf*)(Arow + 16);   // K = kA+16..23    (16B load)
        v16bf bfr[NT];
#pragma unroll
        for (int j = 0; j < NT; ++j)
            bfr[j] = Bbase[((size_t)j * KC + kc) * 32];  // 32B packed fragment

        v16bf a;
#pragma unroll
        for (int i = 0; i < 8; ++i) { a[i] = lo[i]; a[8 + i] = hi[i]; }

        // --- WMMA chain consumes the preloaded fragments ---
#pragma unroll
        for (int j = 0; j < NT; ++j)
            acc[j] = __builtin_amdgcn_wmma_f32_16x16x32_bf16(
                         false, a, false, bfr[j], (short)0, acc[j], false, false);
        Arow += 32;
    }

    // C/D layout: VGPR r -> M = tile_m*16 + half*8 + r, N = ntile*16 + (lane&15)
    const int row0 = tile_m * 16 + half * 8;
#pragma unroll
    for (int j = 0; j < NT; ++j) {
        const int col = (tile_ng * NT + j) * 16 + (lane & 15);
        const float bv = bias ? bias[col] : 0.0f;
#pragma unroll
        for (int r = 0; r < 8; ++r)
            C[(size_t)(row0 + r) * N + col] = acc[j][r] + bv;
    }
}

// ---------------- edge scatter-add: AGG[dst] += G[src], one wave per edge ------
__global__ void scatter_add(const int* __restrict__ src, const int* __restrict__ dst,
                            const float* __restrict__ G, float* __restrict__ AGG,
                            int E, int C) {
    int wave = (blockIdx.x * blockDim.x + threadIdx.x) >> 5;
    int lane = threadIdx.x & 31;
    if (wave >= E) return;
    int s = src[wave], d = dst[wave];
    const float* __restrict__ g = G   + (size_t)s * C;
    float*       __restrict__ o = AGG + (size_t)d * C;
#pragma unroll
    for (int c = lane; c < C; c += 32)
        atomicAdd(&o[c], g[c]);                // coalesced global_atomic_add_f32
}

// ---------------- fused post-aggregation passes ----------------

__device__ __forceinline__ uint32_t hashu(uint32_t x) {
    x ^= x >> 16; x *= 0x7feb352du;
    x ^= x >> 15; x *= 0x846ca68bu;
    x ^= x >> 16; return x;
}

// layer-1 epilogue: norm + bias + relu + dropout(0.5) + next-layer pre-scale -> bf16
__global__ void post1_kernel(const float* __restrict__ AGG,
                             const float* __restrict__ inv_in,
                             const float* __restrict__ inv_out,
                             const float* __restrict__ b1,
                             bf16_t* __restrict__ Hb, int total) {
    int i = blockIdx.x * blockDim.x + threadIdx.x;
    if (i >= total) return;
    int n = i >> 8;            // C = 256
    int c = i & 255;
    float v = AGG[i] * inv_in[n] + b1[c];
    v = fmaxf(v, 0.0f);                          // relu
    v = (hashu((uint32_t)i ^ 0x9E3779B9u) & 1u)  // deterministic 0.5 dropout
            ? v * 2.0f : 0.0f;
    Hb[i] = (bf16_t)(v * inv_out[n]);            // pre-scale for layer-2 GEMM
}

// layer-2 epilogue: norm + bias -> bf16 (feeds the linear head)
__global__ void post2_kernel(const float* __restrict__ AGG,
                             const float* __restrict__ inv_in,
                             const float* __restrict__ b2,
                             bf16_t* __restrict__ Hb, int total) {
    int i = blockIdx.x * blockDim.x + threadIdx.x;
    if (i >= total) return;
    int n = i >> 8;
    int c = i & 255;
    Hb[i] = (bf16_t)(AGG[i] * inv_in[n] + b2[c]);
}

// ---------------------------------------------------------------------------

static inline int cdiv(long long a, long long b) { return (int)((a + b - 1) / b); }

extern "C" void kernel_launch(void* const* d_in, const int* in_sizes, int n_in,
                              void* d_out, int out_size, void* d_ws, size_t ws_size,
                              hipStream_t stream) {
    const float* X   = (const float*)d_in[0];
    const int*   src = (const int*)  d_in[1];
    const int*   dst = (const int*)  d_in[2];
    const float* W1  = (const float*)d_in[3];
    const float* b1  = (const float*)d_in[4];
    const float* W2  = (const float*)d_in[5];
    const float* b2  = (const float*)d_in[6];
    const float* W3  = (const float*)d_in[7];
    const float* b3  = (const float*)d_in[8];
    float* out = (float*)d_out;

    const int M = in_sizes[0] / IN_SZ;   // 100000 nodes
    const int E = in_sizes[1];           // 1600000 edges

    // -------- workspace carve-out (256B aligned) --------
    char* p = (char*)d_ws;
    auto alloc = [&](size_t bytes) -> void* {
        void* r = (void*)p;
        p += (bytes + 255) & ~(size_t)255;
        return r;
    };
    float*  inv_out = (float*) alloc((size_t)M * 4);                 // Dout^-1/2
    float*  inv_in  = (float*) alloc((size_t)M * 4);                 // Din^-1/2
    bf16_t* Xb      = (bf16_t*)alloc((size_t)M * IN_SZ * 2);         // bf16 feats
    bf16_t* W1p     = (bf16_t*)alloc((size_t)IN_SZ  * HID_SZ * 2);
    bf16_t* W2p     = (bf16_t*)alloc((size_t)HID_SZ * HID_SZ * 2);
    bf16_t* W3p     = (bf16_t*)alloc((size_t)HID_SZ * OUT_SZ * 2);
    float*  G       = (float*) alloc((size_t)M * HID_SZ * 4);        // GEMM out
    float*  AGG     = (float*) alloc((size_t)M * HID_SZ * 4);        // scatter acc
    bf16_t* Hb      = Xb;  // reuse feature buffer for bf16 hidden activations
    (void)ws_size; (void)n_in; (void)out_size;

    const int B = 256;

    // -------- degrees & normalization --------
    zero_f32<<<cdiv(2 * M, B), B, 0, stream>>>(inv_out, 2 * M); // inv_out+inv_in contiguous
    degree_kernel<<<cdiv(E, B), B, 0, stream>>>(src, dst, inv_out, inv_in, E);
    finalize_deg<<<cdiv(2 * M, B), B, 0, stream>>>(inv_out, 2 * M);

    // -------- prepare bf16 operands --------
    scale_cvt_bf16<<<cdiv((long long)M * IN_SZ, B), B, 0, stream>>>(X, inv_out, Xb,
                                                                    M * IN_SZ, 9);
    pack_wmma_b<<<cdiv(IN_SZ  * HID_SZ, B), B, 0, stream>>>(W1, W1p, IN_SZ,  HID_SZ);
    pack_wmma_b<<<cdiv(HID_SZ * HID_SZ, B), B, 0, stream>>>(W2, W2p, HID_SZ, HID_SZ);
    pack_wmma_b<<<cdiv(HID_SZ * OUT_SZ, B), B, 0, stream>>>(W3, W3p, HID_SZ, OUT_SZ);

    const int WPB = 128 / 32;  // 4 waves / block

    // -------- layer 1: transform -> aggregate -> epilogue --------
    {
        int tiles = (M / 16) * (HID_SZ / 64);   // NT = 4: 16x64 strips
        gemm_wmma_bf16<4><<<cdiv(tiles, WPB), 128, 0, stream>>>(Xb, W1p, G, nullptr,
                                                                M, HID_SZ, IN_SZ);
    }
    zero_f32<<<cdiv((long long)M * HID_SZ, B), B, 0, stream>>>(AGG, M * HID_SZ);
    scatter_add<<<cdiv((long long)E * 32, B), B, 0, stream>>>(src, dst, G, AGG, E, HID_SZ);
    post1_kernel<<<cdiv((long long)M * HID_SZ, B), B, 0, stream>>>(AGG, inv_in, inv_out,
                                                                   b1, Hb, M * HID_SZ);

    // -------- layer 2 --------
    {
        int tiles = (M / 16) * (HID_SZ / 64);
        gemm_wmma_bf16<4><<<cdiv(tiles, WPB), 128, 0, stream>>>(Hb, W2p, G, nullptr,
                                                                M, HID_SZ, HID_SZ);
    }
    zero_f32<<<cdiv((long long)M * HID_SZ, B), B, 0, stream>>>(AGG, M * HID_SZ);
    scatter_add<<<cdiv((long long)E * 32, B), B, 0, stream>>>(src, dst, G, AGG, E, HID_SZ);
    post2_kernel<<<cdiv((long long)M * HID_SZ, B), B, 0, stream>>>(AGG, inv_in,
                                                                   b2, Hb, M * HID_SZ);

    // -------- linear head: out = Hb @ W3 + b3 (bias fused in WMMA epilogue) ----
    {
        int tiles = (M / 16) * (OUT_SZ / 16);   // NT = 1
        gemm_wmma_bf16<1><<<cdiv(tiles, WPB), 128, 0, stream>>>(Hb, W3p, out, b3,
                                                                M, OUT_SZ, HID_SZ);
    }
}
